// CausalCrossAttention_68745246540015
// MI455X (gfx1250) — matrix operand order repaired and checked by reading, not compiled, not run on hardware
//
#include <hip/hip_runtime.h>
#include <hip/hip_bf16.h>

// Sizes (hardcoded from reference): B=2, N=M=2048, C=1024, H=16, D=64
#define SEQ   2048
#define CH    1024
#define HEADS 16
#define HD    64
#define NEG_INF (-3.0e38f)

typedef __attribute__((ext_vector_type(16))) __bf16 bf16x16;
typedef __attribute__((ext_vector_type(2)))  __bf16 bf16x2;
typedef __attribute__((ext_vector_type(8)))  float  f32x8;
typedef __attribute__((ext_vector_type(2)))  float  f32x2;
typedef __attribute__((ext_vector_type(4)))  int    i32x4;
typedef __attribute__((ext_vector_type(8)))  int    i32x8;
typedef __attribute__((ext_vector_type(4)))  unsigned int u32x4;

union FragAB { bf16x16 v; uint4 q[2]; };

// hardware bf16 converts (v_cvt_pk_bf16_f32 on gfx1250 — confirmed)
__device__ __forceinline__ unsigned int pack2(float lo, float hi) {
  f32x2 f; f.x = lo; f.y = hi;
  bf16x2 b = __builtin_convertvector(f, bf16x2);
  return __builtin_bit_cast(unsigned int, b);
}
__device__ __forceinline__ unsigned short f2bf(float f) {
  __bf16 b = (__bf16)f;
  return __builtin_bit_cast(unsigned short, b);
}

// ---- gfx1250 async global->LDS copy (ASYNCcnt-tracked), guarded fallback ----
#if __has_builtin(__builtin_amdgcn_global_load_async_to_lds_b128) && \
    __has_builtin(__builtin_amdgcn_s_wait_asynccnt)
#define USE_ASYNC_LDS 1
#else
#define USE_ASYNC_LDS 0
#endif

typedef __attribute__((address_space(1))) i32x4* gas_v4i;
typedef __attribute__((address_space(3))) i32x4* las_v4i;

__device__ __forceinline__ void async_cp16(unsigned short* lds,
                                           const unsigned short* g) {
#if USE_ASYNC_LDS
  __builtin_amdgcn_global_load_async_to_lds_b128(
      (gas_v4i)(unsigned short*)g, (las_v4i)lds, 0, 0);
#else
  *(uint4*)lds = *(const uint4*)g;
#endif
}
__device__ __forceinline__ void async_wait() {
#if USE_ASYNC_LDS
  __builtin_amdgcn_s_wait_asynccnt(0);
#endif
}

// ---- gfx1250 Tensor Data Mover: one-instruction 2-D tile DMA to LDS -------
#if __has_builtin(__builtin_amdgcn_tensor_load_to_lds) && \
    __has_builtin(__builtin_amdgcn_s_wait_tensorcnt)
#define USE_TDM 1
#else
#define USE_TDM 0
#endif

__device__ __forceinline__ unsigned lds_off(void* p) {
  return (unsigned)(size_t)(__attribute__((address_space(3))) void*)p;
}

#if USE_TDM
// Load a [tile_rows x 32] bf16 tile (row stride = stride_elems) into LDS with
// 16B padding after each 64B row (-> 80B LDS row stride), per D# spec §8.3-8.5.
__device__ __forceinline__ void tdm_load_tile(unsigned lds_addr,
                                              const unsigned short* gaddr,
                                              int tile_rows, int stride_elems) {
  const unsigned long long ga = (unsigned long long)(size_t)gaddr;
  u32x4 g0 = {};
  g0.x = 1u;                                                // count=1 (valid)
  g0.y = lds_addr;                                          // LDS dest (bytes)
  g0.z = (unsigned)ga;                                      // global_addr[31:0]
  g0.w = (unsigned)((ga >> 32) & 0x01FFFFFFu) | (2u << 30); // [56:32] | type=2
  i32x8 g1 = {};
  g1[0] = (1 << 16)       // data_size = 2 bytes
        | (1 << 20)       // pad_enable
        | (3 << 22)       // pad_interval code 3 = 16 DWORDs (one 64B row)
        | (3 << 25);      // pad_amount  code 3 = 4 DWORDs (16B pad)
  g1[1] = 0;                                  // barrier addr 0; dim0[15:0]=0
  g1[2] = 0x0010;                             // tensor_dim0 = 1<<20 (hi16)
  g1[3] = 0x0010 | (32 << 16);                // tensor_dim1 = 1<<20; tile_dim0=32
  g1[4] = tile_rows;                          // tile_dim1; tile_dim2=0 (2-D)
  g1[5] = stride_elems;                       // tensor_dim0_stride[31:0]
  g1[6] = 0;
  g1[7] = 0;
  i32x4 g2 = {}, g3 = {};
#if defined(__clang_major__) && (__clang_major__ >= 23)
  i32x8 g4 = {};
  __builtin_amdgcn_tensor_load_to_lds(g0, g1, g2, g3, g4, 0);
#else
  __builtin_amdgcn_tensor_load_to_lds(g0, g1, g2, g3, 0);
#endif
}
#endif

// ---------------------------------------------------------------------------
// Pre-pass 1: elementwise f32 -> bf16 (x, context). 8 elems/thread.
// ---------------------------------------------------------------------------
__global__ void cvt_bf16_kernel(const float* __restrict__ in,
                                unsigned short* __restrict__ out) {
  const int i = (blockIdx.x * 256 + threadIdx.x) * 8;
  const float4 a = *(const float4*)(in + i);
  const float4 b = *(const float4*)(in + i + 4);
  uint4 w;
  w.x = pack2(a.x, a.y); w.y = pack2(a.z, a.w);
  w.z = pack2(b.x, b.y); w.w = pack2(b.z, b.w);
  *(uint4*)(out + i) = w;
}

// ---------------------------------------------------------------------------
// Pre-pass 2: W f32 [K=1024][Ncols] -> Wt bf16 [Ncols][K] (transpose+convert),
// 64x64 tiles through LDS so both global phases are coalesced.
// ---------------------------------------------------------------------------
__global__ void wtrans_kernel(const float* __restrict__ W,
                              unsigned short* __restrict__ Wt, int Ncols) {
  __shared__ __align__(16) float sT[64 * 65];
  const int tid = threadIdx.x;
  const int k0 = blockIdx.x * 64;
  const int n0 = blockIdx.y * 64;
  {
    const int nn = tid & 63, kk0 = tid >> 6;
#pragma unroll
    for (int r = 0; r < 16; ++r) {
      const int kk = kk0 + r * 4;
      sT[kk * 65 + nn] = W[(long)(k0 + kk) * Ncols + (n0 + nn)];
    }
  }
  __syncthreads();
  {
    const int kp = tid & 31, nr0 = tid >> 5;
#pragma unroll
    for (int r = 0; r < 8; ++r) {
      const int n = nr0 + r * 8;
      const unsigned int w = pack2(sT[(2 * kp) * 65 + n], sT[(2 * kp + 1) * 65 + n]);
      *(unsigned int*)&Wt[(long)(n0 + n) * CH + k0 + 2 * kp] = w;
    }
  }
}

// ---------------------------------------------------------------------------
// Tiled GEMM core (all-bf16): block tile 128x128, 256 thr = 8 waves.
// Wave (wm,wn) -> 32x64 strip: 8 wmma per 32-k step from 2 A + 4 B frags.
// Double-buffered LDS: A tile via per-thread async b128 copies, B tile via a
// single TDM tensor_load_to_lds (wave 0), both overlapping the wmma block.
// k-loop NOT unrolled so accumulators stay pinned across the backedge.
// ---------------------------------------------------------------------------
template <typename EPI>
__device__ __forceinline__ void gemm_core(const unsigned short* __restrict__ A,
                                          int lda,
                                          const unsigned short* __restrict__ Wt,
                                          int Ktot, EPI epi) {
  __shared__ __align__(16) unsigned short sA[2][128 * 40];   // padded 80B rows
  __shared__ __align__(16) unsigned short sB[2][128 * 40];
  const int tid  = threadIdx.x;
  const int lane = tid & 31;
  const int wid  = tid >> 5;
  const int wm   = wid & 3;        // 4 row strips of 32
  const int wn   = wid >> 2;       // 2 col strips of 64
  const int half = lane >> 4;
  const int l15  = lane & 15;
  const int m0   = blockIdx.y * 128;
  const int n0   = blockIdx.x * 128;

  auto stage = [&](int buf, int k0) {
    {  // A tile 128x32: 512 b128 async copies (2/thread)
      const int row = tid >> 1, ch0 = (tid & 1) * 2;
#pragma unroll
      for (int c = 0; c < 2; ++c) {
        const int ch = ch0 + c;
        async_cp16(&sA[buf][row * 40 + ch * 8],
                   A + (long)(m0 + row) * lda + k0 + ch * 8);
      }
    }
#if USE_TDM
    // B tile 128x32: one TDM descriptor, issued by wave 0 only
    if (wid == 0)
      tdm_load_tile(lds_off(&sB[buf][0]), Wt + (long)n0 * CH + k0, 128, CH);
#else
    {  // fallback: per-thread async copies
      const int row = tid >> 1, ch0 = (tid & 1) * 2;
#pragma unroll
      for (int c = 0; c < 2; ++c) {
        const int ch = ch0 + c;
        async_cp16(&sB[buf][row * 40 + ch * 8],
                   Wt + (long)(n0 + row) * CH + k0 + ch * 8);
      }
    }
#endif
  };

  f32x8 acc[2][4] = {};
  stage(0, 0);
  const int nk = Ktot >> 5;
#pragma unroll 1
  for (int kb = 0; kb < nk; ++kb) {
    const int cur = kb & 1;
    async_wait();                       // my A copies into `cur` complete
#if USE_TDM
    if (wid == 0) __builtin_amdgcn_s_wait_tensorcnt(0);  // B DMA complete
#endif
    __syncthreads();     // everyone's copies done + everyone done reading cur^1
    if (kb + 1 < nk) stage(cur ^ 1, (kb + 1) * 32);  // overlaps the wmma below
    FragAB a[2];  // A 16x32: elem j -> k = half*8+j (j<8), 16+half*8+(j-8)
#pragma unroll
    for (int u = 0; u < 2; ++u) {
      const unsigned short* p =
          &sA[cur][(wm * 32 + u * 16 + l15) * 40 + half * 8];
      a[u].q[0] = *(const uint4*)(p);
      a[u].q[1] = *(const uint4*)(p + 16);
    }
#pragma unroll
    for (int t = 0; t < 4; ++t) {
      FragAB b;   // B 32x16: lane n = l15, elem j -> k = half*8+j (j<8), 16+half*8+(j-8)
      const unsigned short* p =
          &sB[cur][(wn * 64 + t * 16 + l15) * 40 + half * 8];
      b.q[0] = *(const uint4*)(p);
      b.q[1] = *(const uint4*)(p + 16);
#pragma unroll
      for (int u = 0; u < 2; ++u)
        acc[u][t] = __builtin_amdgcn_wmma_f32_16x16x32_bf16(
            false, a[u].v, false, b.v, (short)0, acc[u][t], false, false);
    }
  }
  // C/D layout: elem r -> row r+8*half, col l15
#pragma unroll
  for (int u = 0; u < 2; ++u)
#pragma unroll
    for (int t = 0; t < 4; ++t)
#pragma unroll
      for (int r = 0; r < 8; ++r)
        epi(m0 + wm * 32 + u * 16 + r + 8 * half,
            n0 + wn * 64 + t * 16 + l15, acc[u][t][r]);
}

// ---------------------- epilogues ----------------------
struct EpiQ {            // Q -> [B,H,N,D] bf16
  unsigned short* Qo;
  __device__ void operator()(int gm, int gc, float v) const {
    const int b = gm >> 11, n = gm & 2047, h = gc >> 6, d = gc & 63;
    Qo[(((long)b * HEADS + h) * SEQ + n) * HD + d] = f2bf(v);
  }
};
struct EpiKV {           // K -> [B,H,M,D], V -> transposed [B,H,D,M]
  unsigned short* Ko; unsigned short* Vt;
  __device__ void operator()(int gm, int gc, float v) const {
    const int b = gm >> 11, m = gm & 2047;
    if (gc < CH) {
      const int h = gc >> 6, d = gc & 63;
      Ko[(((long)b * HEADS + h) * SEQ + m) * HD + d] = f2bf(v);
    } else {
      const int c = gc - CH, h = c >> 6, d = c & 63;
      Vt[(((long)b * HEADS + h) * HD + d) * SEQ + m] = f2bf(v);
    }
  }
};
struct EpiOut {          // final fp32 output + bias
  float* out; const float* bias;
  __device__ void operator()(int gm, int gc, float v) const {
    out[(long)gm * CH + gc] = v + bias[gc];
  }
};

__global__ void proj_q_kernel(const unsigned short* __restrict__ xb,
                              const unsigned short* __restrict__ Wqt,
                              unsigned short* __restrict__ Q) {
  gemm_core(xb, CH, Wqt, CH, EpiQ{Q});
}
__global__ void proj_kv_kernel(const unsigned short* __restrict__ cb,
                               const unsigned short* __restrict__ Wkvt,
                               unsigned short* __restrict__ K,
                               unsigned short* __restrict__ Vt) {
  gemm_core(cb, CH, Wkvt, CH, EpiKV{K, Vt});
}
__global__ void proj_out_kernel(const unsigned short* __restrict__ O,
                                const unsigned short* __restrict__ Wpt,
                                const float* __restrict__ bias,
                                float* __restrict__ out) {
  gemm_core(O, CH, Wpt, CH, EpiOut{out, bias});
}

// ---------------------------------------------------------------------------
// Flash attention: one wave owns a 16-query tile, streams 32-key chunks.
// S = Q*K^T via 4 wmma, online softmax via 16-lane shfl_xor reductions,
// P repacked C-layout -> A-layout through per-wave LDS, O += P*V via 4 wmma.
// global_prefetch_b8 pulls the next K/V chunk through L2.
// ---------------------------------------------------------------------------
__global__ void attn_kernel(const unsigned short* __restrict__ Q,
                            const unsigned short* __restrict__ K,
                            const unsigned short* __restrict__ Vt,
                            unsigned short* __restrict__ O) {
  __shared__ __align__(16) unsigned short sP[8][16 * 40];  // per-wave P tile
  const int tid  = threadIdx.x;
  const int lane = tid & 31;
  const int wid  = tid >> 5;
  const int half = lane >> 4;
  const int l15  = lane & 15;
  const int bh   = blockIdx.y;           // b*H + h
  const int b    = bh >> 4, h = bh & 15;
  const int q0   = (blockIdx.x * 8 + wid) * 16;

  const unsigned short* Qb = Q  + (long)bh * SEQ * HD;
  const unsigned short* Kb = K  + (long)bh * SEQ * HD;
  const unsigned short* Vb = Vt + (long)bh * HD * SEQ;

  // Q fragments for d=[0,32) and [32,64)
  FragAB qf[2];
  {
    const unsigned short* qr = Qb + (long)(q0 + l15) * HD;
#pragma unroll
    for (int kt = 0; kt < 2; ++kt) {
      const unsigned short* p = qr + kt * 32 + half * 8;
      qf[kt].q[0] = *(const uint4*)(p);
      qf[kt].q[1] = *(const uint4*)(p + 16);
    }
  }

  f32x8 o[4] = {};
  float mi[8], li[8];
#pragma unroll
  for (int r = 0; r < 8; ++r) { mi[r] = NEG_INF; li[r] = 0.f; }

  const int jend = q0 + 16;              // causal: keys <= query
#pragma unroll 1
  for (int j0 = 0; j0 < jend; j0 += 32) {
    if (j0 + 32 < jend) {                // prefetch next chunk (K rows + V rows)
      __builtin_prefetch(Kb + (long)(j0 + 32 + lane) * HD, 0, 3);
      __builtin_prefetch(Vb + (long)(lane * 2) * SEQ + j0 + 32, 0, 3);
    }
    f32x8 s[2] = {};
#pragma unroll
    for (int st = 0; st < 2; ++st) {
      const unsigned short* kr = Kb + (long)(j0 + st * 16 + l15) * HD;
#pragma unroll
      for (int kt = 0; kt < 2; ++kt) {
        FragAB kf;  // K row as B fragment (B[k][n] = K[n][k])
        const unsigned short* p = kr + kt * 32 + half * 8;
        kf.q[0] = *(const uint4*)(p);
        kf.q[1] = *(const uint4*)(p + 16);
        s[st] = __builtin_amdgcn_wmma_f32_16x16x32_bf16(
            false, qf[kt].v, false, kf.v, (short)0, s[st], false, false);
      }
    }
    // scale, causal mask, online softmax (row = r + 8*half, key col = l15)
    float pr[2][8];
#pragma unroll
    for (int r = 0; r < 8; ++r) {
      const int qrow = q0 + r + 8 * half;
      float v0 = s[0][r] * 0.125f;               // D^-0.5
      float v1 = s[1][r] * 0.125f;
      if (j0 + l15 > qrow)      v0 = NEG_INF;
      if (j0 + 16 + l15 > qrow) v1 = NEG_INF;
      float vmax = fmaxf(v0, v1);
#pragma unroll
      for (int msk = 1; msk < 16; msk <<= 1)
        vmax = fmaxf(vmax, __shfl_xor(vmax, msk, 32));
      const float newm  = fmaxf(mi[r], vmax);
      const float alpha = __expf(mi[r] - newm);
      const float p0 = __expf(v0 - newm);
      const float p1 = __expf(v1 - newm);
      float ps = p0 + p1;
#pragma unroll
      for (int msk = 1; msk < 16; msk <<= 1)
        ps += __shfl_xor(ps, msk, 32);
      li[r] = li[r] * alpha + ps;
      mi[r] = newm;
#pragma unroll
      for (int t = 0; t < 4; ++t) o[t][r] *= alpha;
      pr[0][r] = p0; pr[1][r] = p1;
    }
    // repack P (C layout) -> A layout via this wave's private LDS tile;
    // same-wave DS ops are in order, no barrier needed.
    unsigned short* Ps = sP[wid];
#pragma unroll
    for (int st = 0; st < 2; ++st)
#pragma unroll
      for (int r = 0; r < 8; ++r)
        Ps[(r + 8 * half) * 40 + st * 16 + l15] = f2bf(pr[st][r]);
    FragAB pf;
    {
      const unsigned short* pp = &Ps[l15 * 40 + half * 8];
      pf.q[0] = *(const uint4*)(pp);
      pf.q[1] = *(const uint4*)(pp + 16);
    }
    // O += P * V  (V^T layout makes B-frag reads contiguous in m)
#pragma unroll
    for (int t = 0; t < 4; ++t) {
      FragAB vf;
      const unsigned short* vp = Vb + (long)(t * 16 + l15) * SEQ + j0 + half * 8;
      vf.q[0] = *(const uint4*)(vp);
      vf.q[1] = *(const uint4*)(vp + 16);
      o[t] = __builtin_amdgcn_wmma_f32_16x16x32_bf16(
          false, pf.v, false, vf.v, (short)0, o[t], false, false);
    }
  }
  // normalize and store O as [B,N,C] bf16 (ready for out-projection GEMM)
#pragma unroll
  for (int t = 0; t < 4; ++t)
#pragma unroll
    for (int r = 0; r < 8; ++r) {
      const int qrow = q0 + r + 8 * half;
      const int d    = t * 16 + l15;
      O[((long)b * SEQ + qrow) * CH + h * HD + d] = f2bf(o[t][r] / li[r]);
    }
}

extern "C" void kernel_launch(void* const* d_in, const int* in_sizes, int n_in,
                              void* d_out, int out_size, void* d_ws, size_t ws_size,
                              hipStream_t stream) {
  const float* x   = (const float*)d_in[0];
  const float* ctx = (const float*)d_in[1];
  // d_in[2] = attn_mask (bool tril) — causal mask is computed analytically
  const float* Wq  = (const float*)d_in[3];
  const float* Wkv = (const float*)d_in[4];
  const float* Wp  = (const float*)d_in[5];
  const float* bp  = (const float*)d_in[6];
  float* out = (float*)d_out;

  char* ws = (char*)d_ws;
  unsigned short* Qb   = (unsigned short*)(ws);                        // 8 MiB
  unsigned short* Kb   = (unsigned short*)(ws + ((size_t)8  << 20));   // 8 MiB
  unsigned short* Vt   = (unsigned short*)(ws + ((size_t)16 << 20));   // 8 MiB
  unsigned short* Ob   = (unsigned short*)(ws + ((size_t)24 << 20));   // 8 MiB
  unsigned short* xb   = (unsigned short*)(ws + ((size_t)32 << 20));   // 8 MiB
  unsigned short* cb   = (unsigned short*)(ws + ((size_t)40 << 20));   // 8 MiB
  unsigned short* Wqt  = (unsigned short*)(ws + ((size_t)48 << 20));   // 2 MiB
  unsigned short* Wkvt = (unsigned short*)(ws + ((size_t)50 << 20));   // 4 MiB
  unsigned short* Wpt  = (unsigned short*)(ws + ((size_t)54 << 20));   // 2 MiB

  const dim3 blk(256);
  // pre-passes: convert activations, transpose+convert weights (bf16)
  cvt_bf16_kernel<<<2048, blk, 0, stream>>>(x, xb);     // 2*2048*1024 / (256*8)
  cvt_bf16_kernel<<<2048, blk, 0, stream>>>(ctx, cb);
  wtrans_kernel<<<dim3(16, 16), blk, 0, stream>>>(Wq, Wqt, CH);
  wtrans_kernel<<<dim3(16, 32), blk, 0, stream>>>(Wkv, Wkvt, 2 * CH);
  wtrans_kernel<<<dim3(16, 16), blk, 0, stream>>>(Wp, Wpt, CH);
  // main pipeline
  proj_q_kernel  <<<dim3(8, 32),  blk, 0, stream>>>(xb, Wqt, Qb);
  proj_kv_kernel <<<dim3(16, 32), blk, 0, stream>>>(cb, Wkvt, Kb, Vt);
  attn_kernel    <<<dim3(16, 32), blk, 0, stream>>>(Qb, Kb, Vt, Ob);
  proj_out_kernel<<<dim3(8, 32),  blk, 0, stream>>>(Ob, Wpt, bp, out);
}
